// CausalCrossAttention_83107617177877
// MI455X (gfx1250) — matrix operand order repaired and checked by reading, not compiled, not run on hardware
//
#include <hip/hip_runtime.h>
#include <math.h>
#include <stdint.h>

#define B_   2
#define C_   512
#define T_   16
#define HW_  1024
#define S_   64
#define D_   1024
#define G_   32
#define CPG_ 16      // channels per group
#define NR_  4       // S_/T_ context frames per x frame

typedef __attribute__((ext_vector_type(16))) __bf16 v16bf;
typedef __attribute__((ext_vector_type(8)))  float  v8f;
typedef __attribute__((ext_vector_type(4)))  float  vf4;
typedef __attribute__((ext_vector_type(4)))  unsigned int v4u;

union Frag { v16bf v; vf4 f4[2]; };
static_assert(sizeof(Frag) == 32, "frag size");

__device__ __forceinline__ uint16_t f2bf(float x) {
  uint32_t u = __float_as_uint(x);
  u += 0x7FFFu + ((u >> 16) & 1u);     // round-to-nearest-even
  return (uint16_t)(u >> 16);
}

__device__ __forceinline__ v8f wmma_bf16(const Frag& a, const Frag& b, v8f c) {
  return __builtin_amdgcn_wmma_f32_16x16x32_bf16(false, a.v, false, b.v, (short)0, c, false, false);
}

// A fragment: 16x32 (MxK) bf16, row-major [M,K], ld in elements.
// lane m = l&15; lanes 0-15 K-base 0, lanes 16-31 K-base 8 (VGPR0-3), +16 (VGPR4-7).
__device__ __forceinline__ Frag load_frag_A(const uint16_t* base, int ld) {
  const int l  = threadIdx.x & 31;
  const int m  = l & 15;
  const int ks = (l >> 4) * 8;
  const uint16_t* r = base + m * ld;
  Frag f;
  f.f4[0] = *reinterpret_cast<const vf4*>(r + ks);
  f.f4[1] = *reinterpret_cast<const vf4*>(r + 16 + ks);
  return f;
}

// B fragment: 32x16 (KxN) bf16 stored as [N,K] row-major (ld in elements).
// lane n = l&15; lanes 0-15 K-base 0, lanes 16-31 K-base 8 (VGPR0-3), +16 (VGPR4-7).
__device__ __forceinline__ Frag load_frag_B(const uint16_t* base, int ld) {
  const int l  = threadIdx.x & 31;
  const int n  = l & 15;
  const int kb = (l >> 4) * 8;
  const uint16_t* r = base + n * ld + kb;
  Frag f;
  f.f4[0] = *reinterpret_cast<const vf4*>(r);
  f.f4[1] = *reinterpret_cast<const vf4*>(r + 16);
  return f;
}

// ---------------- fp32 -> bf16 weight conversion ----------------
__global__ void cvt_bf16_kernel(const float* __restrict__ src,
                                uint16_t* __restrict__ dst, int n) {
  int i = blockIdx.x * blockDim.x + threadIdx.x;
  if (i < n) dst[i] = f2bf(src[i]);
}

// ---------------- GroupNorm: x[b,c,t,h,w] -> h_bf16[b,t,p,c] ----------------
__global__ void gn_kernel(const float* __restrict__ x, const float* __restrict__ gamma,
                          const float* __restrict__ beta, uint16_t* __restrict__ h) {
  const int bid = blockIdx.x;            // b*G*T + g*T + t
  const int t = bid % T_;
  const int g = (bid / T_) % G_;
  const int b = bid / (T_ * G_);
  const int tid = threadIdx.x;
  const int N = CPG_ * HW_;              // 16384 elements per (b,g,t)

  float s = 0.f, ss = 0.f;
  for (int idx = tid; idx < N; idx += 256) {
    int c = idx >> 10;                   // /HW_
    int p = idx & (HW_ - 1);
    float xv = x[((b * C_ + g * CPG_ + c) * T_ + t) * HW_ + p];
    s += xv; ss += xv * xv;
  }
  #pragma unroll
  for (int m = 16; m >= 1; m >>= 1) {
    s  += __shfl_xor(s,  m, 32);
    ss += __shfl_xor(ss, m, 32);
  }
  __shared__ float rs[8], rss[8];
  __shared__ float smean, srstd;
  const int wv = tid >> 5;
  if ((tid & 31) == 0) { rs[wv] = s; rss[wv] = ss; }
  __syncthreads();
  if (tid == 0) {
    float S = 0.f, SS = 0.f;
    #pragma unroll
    for (int i = 0; i < 8; ++i) { S += rs[i]; SS += rss[i]; }
    float mean = S / (float)N;
    float var  = SS / (float)N - mean * mean;
    smean = mean;
    srstd = rsqrtf(var + 1e-5f);
  }
  __syncthreads();
  const float mean = smean, rstd = srstd;
  for (int idx = tid; idx < N; idx += 256) {
    int c  = idx >> 10;
    int p  = idx & (HW_ - 1);
    int ch = g * CPG_ + c;
    float xv = x[((b * C_ + ch) * T_ + t) * HW_ + p];
    float hv = (xv - mean) * rstd * gamma[ch] + beta[ch];
    h[((b * T_ + t) * HW_ + p) * C_ + ch] = f2bf(hv);
  }
}

// ---------------- kv projection (scalar fp32, 0.27 GFLOP) ----------------
// k_bf16[b,s,c] ; v stored transposed vT_bf16[b,c,s]
__global__ void kv_kernel(const float* __restrict__ ctx, const float* __restrict__ wkv,
                          const float* __restrict__ bkv,
                          uint16_t* __restrict__ kbuf, uint16_t* __restrict__ vbuf) {
  int i  = blockIdx.x * blockDim.x + threadIdx.x;  // over B*S*2C
  int oc = i % (2 * C_);
  int s  = (i / (2 * C_)) % S_;
  int b  = i / (2 * C_ * S_);
  const float* cr = ctx + (b * S_ + s) * D_;
  const float* wr = wkv + oc * D_;
  float acc = bkv[oc];
  for (int d = 0; d < D_; ++d) acc = fmaf(cr[d], wr[d], acc);
  if (oc < C_) kbuf[(b * S_ + s) * C_ + oc]        = f2bf(acc);
  else         vbuf[(b * C_ + (oc - C_)) * S_ + s] = f2bf(acc);
}

// ---------------- q projection: q[b,t,p,o] = (wq x h + bq) * C^-0.5 ----------------
// per (b,t): M = o (512), N = p (1024), K = c (512). A = wq rows, B = h[bt,p,c].
__global__ void qproj_kernel(const uint16_t* __restrict__ h, const uint16_t* __restrict__ wq,
                             const float* __restrict__ bq, uint16_t* __restrict__ q) {
  const int wv  = threadIdx.x >> 5;
  const int l   = threadIdx.x & 31;
  const int job = blockIdx.x * 8 + wv;   // 16384 = 32 bt * 32 mt * 16 ng
  const int ng  = job & 15;
  const int mt  = (job >> 4) & 31;
  const int bt  = job >> 9;

  v8f acc[4] = {};
  for (int kc = 0; kc < C_; kc += 32) {
    Frag a = load_frag_A(wq + mt * 16 * C_ + kc, C_);
    #pragma unroll
    for (int j = 0; j < 4; ++j) {
      int nt = ng * 4 + j;
      Frag bf = load_frag_B(h + (bt * HW_ + nt * 16) * C_ + kc, C_);
      acc[j] = wmma_bf16(a, bf, acc[j]);
    }
  }
  const float scale = 0.04419417382415922f;   // 1/sqrt(512)
  const int half  = l >> 4;
  const int obase = mt * 16 + half * 8;       // output channel base for this lane's rows
  const vf4 bql = *reinterpret_cast<const vf4*>(bq + obase);
  const vf4 bqh = *reinterpret_cast<const vf4*>(bq + obase + 4);
  float bias[8] = {bql[0], bql[1], bql[2], bql[3], bqh[0], bqh[1], bqh[2], bqh[3]};
  #pragma unroll
  for (int j = 0; j < 4; ++j) {
    int p = (ng * 4 + j) * 16 + (l & 15);
    uint16_t* dst = q + (bt * HW_ + p) * C_ + obase;
    v4u pk;
    #pragma unroll
    for (int e = 0; e < 4; ++e) {
      float lo = (acc[j][2 * e]     + bias[2 * e])     * scale;
      float hi = (acc[j][2 * e + 1] + bias[2 * e + 1]) * scale;
      pk[e] = (uint32_t)f2bf(lo) | ((uint32_t)f2bf(hi) << 16);
    }
    *reinterpret_cast<v4u*>(dst) = pk;
  }
}

// ---------------- fused block-causal attention, one wave per (b, p) ----------------
__global__ void attn_kernel(const uint16_t* __restrict__ q, const uint16_t* __restrict__ kbuf,
                            const uint16_t* __restrict__ vbuf, uint16_t* __restrict__ obuf) {
  __shared__ uint16_t wlds[8][T_ * S_];   // per-wave 16x64 softmax weights (bf16)
  const int wv  = threadIdx.x >> 5;
  const int l   = threadIdx.x & 31;
  const int job = blockIdx.x * 8 + wv;    // B*HW = 2048
  const int p   = job & (HW_ - 1);
  const int b   = job >> 10;
  const int half = l >> 4;
  const int lm   = l & 15;

  // scores[16 x 64] = q[16 x 512] . k^T, 4 N-tiles over s
  v8f sc[4] = {};
  const uint16_t* qbase = q + (b * T_ * HW_ + p) * C_;   // rows t, ld = HW_*C_
  for (int kc = 0; kc < C_; kc += 32) {
    Frag a = load_frag_A(qbase + kc, HW_ * C_);
    #pragma unroll
    for (int st = 0; st < 4; ++st) {
      Frag bf = load_frag_B(kbuf + (b * S_ + st * 16) * C_ + kc, C_);
      sc[st] = wmma_bf16(a, bf, sc[st]);
    }
  }
  // block-causal mask: row t attends s < (t+1)*4
  #pragma unroll
  for (int st = 0; st < 4; ++st) {
    int s = st * 16 + lm;
    #pragma unroll
    for (int e = 0; e < 8; ++e) {
      int t = half * 8 + e;
      if (s >= (t + 1) * NR_) sc[st][e] = -INFINITY;
    }
  }
  // softmax over s: per row t = half*8+e; reduce over 4 tiles + 16 lanes of same half
  float rinv[8];
  #pragma unroll
  for (int e = 0; e < 8; ++e) {
    float m = fmaxf(fmaxf(sc[0][e], sc[1][e]), fmaxf(sc[2][e], sc[3][e]));
    #pragma unroll
    for (int msk = 8; msk >= 1; msk >>= 1) m = fmaxf(m, __shfl_xor(m, msk, 32));
    #pragma unroll
    for (int st = 0; st < 4; ++st) sc[st][e] = __expf(sc[st][e] - m);
    float sm = sc[0][e] + sc[1][e] + sc[2][e] + sc[3][e];
    #pragma unroll
    for (int msk = 8; msk >= 1; msk >>= 1) sm += __shfl_xor(sm, msk, 32);
    rinv[e] = 1.0f / sm;
  }
  // transpose w into A-fragment layout via LDS: wlds[t][s] (bf16)
  #pragma unroll
  for (int st = 0; st < 4; ++st) {
    int s = st * 16 + lm;
    #pragma unroll
    for (int e = 0; e < 8; ++e)
      wlds[wv][(half * 8 + e) * S_ + s] = f2bf(sc[st][e] * rinv[e]);
  }
  __builtin_amdgcn_wave_barrier();
  asm volatile("s_wait_dscnt 0" ::: "memory");
  Frag wa0 = load_frag_A(&wlds[wv][0],  S_);   // K = s 0..31
  Frag wa1 = load_frag_A(&wlds[wv][32], S_);   // K = s 32..63

  // o[16 x C] = w[16 x 64] . v[64 x C]; vT layout [b,c,s] is the B-fragment layout
  for (int ct = 0; ct < C_ / 16; ++ct) {
    v8f acc = {};
    Frag vb0 = load_frag_B(vbuf + (b * C_ + ct * 16) * S_,      S_);
    Frag vb1 = load_frag_B(vbuf + (b * C_ + ct * 16) * S_ + 32, S_);
    acc = wmma_bf16(wa0, vb0, acc);
    acc = wmma_bf16(wa1, vb1, acc);
    int c = ct * 16 + lm;
    #pragma unroll
    for (int e = 0; e < 8; ++e) {
      int t = half * 8 + e;
      obuf[((b * HW_ + p) * T_ + t) * C_ + c] = f2bf(acc[e]);
    }
  }
}

// ---------------- o projection + residual: out = x + o.wo^T + bo ----------------
// per (b,t): M = p (1024), N = co (512), K = c (512). A = o[b,p,t,c], B = wo rows.
__global__ void oproj_kernel(const uint16_t* __restrict__ obuf, const uint16_t* __restrict__ wo,
                             const float* __restrict__ bo, const float* __restrict__ x,
                             float* __restrict__ out) {
  const int wv  = threadIdx.x >> 5;
  const int l   = threadIdx.x & 31;
  const int job = blockIdx.x * 8 + wv;   // 16384 = 32 bt * 64 mt * 8 ng
  const int ng  = job & 7;
  const int mt  = (job >> 3) & 63;
  const int bt  = job >> 9;
  const int b   = bt / T_, t = bt % T_;

  v8f acc[4] = {};
  const uint16_t* abase = obuf + ((b * HW_ + mt * 16) * T_ + t) * C_;  // rows p, ld = T_*C_
  for (int kc = 0; kc < C_; kc += 32) {
    Frag a = load_frag_A(abase + kc, T_ * C_);
    #pragma unroll
    for (int j = 0; j < 4; ++j) {
      int nt = ng * 4 + j;
      Frag bf = load_frag_B(wo + nt * 16 * C_ + kc, C_);
      acc[j] = wmma_bf16(a, bf, acc[j]);
    }
  }
  const int half = l >> 4;
  const int p0 = mt * 16 + half * 8;     // 8 consecutive p per lane
  #pragma unroll
  for (int j = 0; j < 4; ++j) {
    int co = (ng * 4 + j) * 16 + (l & 15);
    float bias = bo[co];
    const float* xr = x   + ((b * C_ + co) * T_ + t) * HW_ + p0;
    float*       orw = out + ((b * C_ + co) * T_ + t) * HW_ + p0;
    vf4 x0 = *reinterpret_cast<const vf4*>(xr);
    vf4 x1 = *reinterpret_cast<const vf4*>(xr + 4);
    vf4 r0, r1;
    #pragma unroll
    for (int e = 0; e < 4; ++e) {
      r0[e] = x0[e] + acc[j][e]     + bias;
      r1[e] = x1[e] + acc[j][e + 4] + bias;
    }
    *reinterpret_cast<vf4*>(orw)     = r0;
    *reinterpret_cast<vf4*>(orw + 4) = r1;
  }
}

extern "C" void kernel_launch(void* const* d_in, const int* in_sizes, int n_in,
                              void* d_out, int out_size, void* d_ws, size_t ws_size,
                              hipStream_t stream) {
  (void)in_sizes; (void)n_in; (void)out_size; (void)ws_size;
  const float* x     = (const float*)d_in[0];
  const float* ctx   = (const float*)d_in[1];
  const float* gamma = (const float*)d_in[2];
  const float* beta  = (const float*)d_in[3];
  const float* wq    = (const float*)d_in[4];
  const float* bq    = (const float*)d_in[5];
  const float* wkv   = (const float*)d_in[6];
  const float* bkv   = (const float*)d_in[7];
  const float* wo    = (const float*)d_in[8];
  const float* bo    = (const float*)d_in[9];
  float* out = (float*)d_out;

  char* ws = (char*)d_ws;
  size_t off = 0;
  auto alloc = [&](size_t bytes) -> char* {
    char* p = ws + off;
    off += (bytes + 255) & ~((size_t)255);
    return p;
  };
  const size_t bigElems = (size_t)B_ * T_ * HW_ * C_;          // 16,777,216
  uint16_t* hbuf = (uint16_t*)alloc(bigElems * 2);             // h; reused as o after qproj
  uint16_t* qbuf = (uint16_t*)alloc(bigElems * 2);
  uint16_t* wqb  = (uint16_t*)alloc((size_t)C_ * C_ * 2);
  uint16_t* wob  = (uint16_t*)alloc((size_t)C_ * C_ * 2);
  uint16_t* kbuf = (uint16_t*)alloc((size_t)B_ * S_ * C_ * 2);
  uint16_t* vbuf = (uint16_t*)alloc((size_t)B_ * S_ * C_ * 2); // transposed [b,c,s]

  cvt_bf16_kernel<<<(C_ * C_) / 256, 256, 0, stream>>>(wq, wqb, C_ * C_);
  cvt_bf16_kernel<<<(C_ * C_) / 256, 256, 0, stream>>>(wo, wob, C_ * C_);
  gn_kernel<<<B_ * G_ * T_, 256, 0, stream>>>(x, gamma, beta, hbuf);
  kv_kernel<<<(B_ * S_ * 2 * C_) / 256, 256, 0, stream>>>(ctx, wkv, bkv, kbuf, vbuf);
  qproj_kernel<<<2048, 256, 0, stream>>>(hbuf, wqb, bq, qbuf);
  attn_kernel<<<(B_ * HW_) / 8, 256, 0, stream>>>(qbuf, kbuf, vbuf, hbuf /* obuf reuse */);
  oproj_kernel<<<2048, 256, 0, stream>>>(hbuf, wob, bo, x, out);
}
